// GQA_47004122087999
// MI455X (gfx1250) — matrix operand order repaired and checked by reading, not compiled, not run on hardware
//
#include <hip/hip_runtime.h>

typedef __bf16 bf16;
typedef __bf16 v16bf __attribute__((ext_vector_type(16)));
typedef float  v8f   __attribute__((ext_vector_type(8)));

#define HIDDEN  2048
#define NHEADS  32
#define NKV     8
#define HD      64
#define SEQ     2048
#define BATCH   2
#define NTOK    (BATCH*SEQ)          /* 4096 */
#define KVDIM   (NKV*HD)             /* 512  */
#define QKVN    (HIDDEN + 2*KVDIM)   /* 3072 */
#define KOFF    HIDDEN               /* 2048 */
#define VOFF    (HIDDEN + KVDIM)     /* 2560 */

#if __has_builtin(__builtin_amdgcn_global_load_async_to_lds_b128) && \
    __has_builtin(__builtin_amdgcn_s_wait_asynccnt)
#define HAVE_ASYNC 1
#else
#define HAVE_ASYNC 0
#endif

// ---------------------------------------------------------------- helpers

// Exact pointee type the async-LDS builtins expect: vector_size(16) int.
typedef int v4i_mem __attribute__((vector_size(16)));
typedef __attribute__((address_space(1))) v4i_mem* gas_b128_t;  // global
typedef __attribute__((address_space(3))) v4i_mem* las_b128_t;  // LDS

__device__ __forceinline__ void cp_async_b128(bf16* lds, const bf16* g) {
#if HAVE_ASYNC
  __builtin_amdgcn_global_load_async_to_lds_b128(
      (gas_b128_t)(g), (las_b128_t)(lds), 0, 0);
#else
  *(uint4*)lds = *(const uint4*)g;
#endif
}

__device__ __forceinline__ void cp_async_wait() {
#if HAVE_ASYNC
  __builtin_amdgcn_s_wait_asynccnt(0);
#endif
}

__device__ __forceinline__ v8f zero8() {
  v8f z;
  #pragma unroll
  for (int i = 0; i < 8; ++i) z[i] = 0.0f;
  return z;
}

__device__ __forceinline__ v8f wmma_bf16(v16bf a, v16bf b, v8f c) {
  // 8 args: (neg_a, A, neg_b, B, c_mod, C, reuse_a, reuse_b)
  return __builtin_amdgcn_wmma_f32_16x16x32_bf16(
      false, a, false, b, (short)0, c, false, false);
}

// 16-lane butterfly reduction, pure VALU via DPP16 (wave32 halves independent).
__device__ __forceinline__ float dpp16_fmax(float x) {
  int t;
  t = __builtin_amdgcn_update_dpp(__float_as_int(x), __float_as_int(x),
                                  0xB1 /*quad_perm(1,0,3,2)*/, 0xF, 0xF, true);
  x = fmaxf(x, __int_as_float(t));
  t = __builtin_amdgcn_update_dpp(__float_as_int(x), __float_as_int(x),
                                  0x4E /*quad_perm(2,3,0,1)*/, 0xF, 0xF, true);
  x = fmaxf(x, __int_as_float(t));
  t = __builtin_amdgcn_update_dpp(__float_as_int(x), __float_as_int(x),
                                  0x141 /*row_half_mirror*/, 0xF, 0xF, true);
  x = fmaxf(x, __int_as_float(t));
  t = __builtin_amdgcn_update_dpp(__float_as_int(x), __float_as_int(x),
                                  0x140 /*row_mirror*/, 0xF, 0xF, true);
  x = fmaxf(x, __int_as_float(t));
  return x;
}

__device__ __forceinline__ float dpp16_fadd(float x) {
  int t;
  t = __builtin_amdgcn_update_dpp(__float_as_int(x), __float_as_int(x),
                                  0xB1, 0xF, 0xF, true);
  x += __int_as_float(t);
  t = __builtin_amdgcn_update_dpp(__float_as_int(x), __float_as_int(x),
                                  0x4E, 0xF, 0xF, true);
  x += __int_as_float(t);
  t = __builtin_amdgcn_update_dpp(__float_as_int(x), __float_as_int(x),
                                  0x141, 0xF, 0xF, true);
  x += __int_as_float(t);
  t = __builtin_amdgcn_update_dpp(__float_as_int(x), __float_as_int(x),
                                  0x140, 0xF, 0xF, true);
  x += __int_as_float(t);
  return x;
}

// A fragment: 16x32 (MxK) tile, row-major source p[m*ld + k].
__device__ __forceinline__ v16bf frag_a(const bf16* p, int ld) {
  const int lane = threadIdx.x & 31;
  const int m  = lane & 15;
  const int hl = lane >> 4;
  v16bf a;
  #pragma unroll
  for (int j = 0; j < 8; ++j) {
    const int k = (hl << 4) + (j << 1);
    a[2*j]   = p[m*ld + k];
    a[2*j+1] = p[m*ld + k + 1];
  }
  return a;
}

// B fragment: 32x16 (KxN), source stored N-major (p[n*ld + k]).
__device__ __forceinline__ v16bf frag_b_nk(const bf16* p, int ld) {
  const int lane = threadIdx.x & 31;
  const int n  = lane & 15;
  const int hl = lane >> 4;
  v16bf b;
  #pragma unroll
  for (int j = 0; j < 8; ++j) {
    const int k = (hl << 4) + (j << 1);
    b[2*j]   = p[n*ld + k];
    b[2*j+1] = p[n*ld + k + 1];
  }
  return b;
}

// B fragment: 32x16 (KxN), source stored K-major (p[k*ld + n]).
__device__ __forceinline__ v16bf frag_b_kn(const bf16* p, int ld) {
  const int lane = threadIdx.x & 31;
  const int n  = lane & 15;
  const int hl = lane >> 4;
  v16bf b;
  #pragma unroll
  for (int j = 0; j < 8; ++j) {
    b[2*j]   = p[(hl*16 + 2*j    )*ld + n];
    b[2*j+1] = p[(hl*16 + 2*j + 1)*ld + n];
  }
  return b;
}

// ---------------------------------------------------------------- convert

__global__ void k_cvt(const float* __restrict__ src, bf16* __restrict__ dst, int n) {
  int i = blockIdx.x * blockDim.x + threadIdx.x;
  if (i < n) dst[i] = (bf16)src[i];
}

__global__ void k_cvt_qkv(const float* __restrict__ wq, const float* __restrict__ wk,
                          const float* __restrict__ wv, bf16* __restrict__ dst) {
  int i = blockIdx.x * blockDim.x + threadIdx.x;
  if (i >= QKVN * HIDDEN) return;
  int n = i / HIDDEN;
  int k = i - n * HIDDEN;
  float v;
  if (n < HIDDEN)              v = wq[(size_t)n * HIDDEN + k];
  else if (n < HIDDEN + KVDIM) v = wk[(size_t)(n - HIDDEN) * HIDDEN + k];
  else                         v = wv[(size_t)(n - HIDDEN - KVDIM) * HIDDEN + k];
  dst[i] = (bf16)v;
}

// ---------------------------------------------------------------- RoPE (in place, Q scaled by 1/sqrt(HD))

__global__ void k_rope(bf16* __restrict__ qkv) {
  int i = blockIdx.x * blockDim.x + threadIdx.x;
  const int total = NTOK * (NHEADS + NKV) * 32;
  if (i >= total) return;
  const int d   = i & 31;
  const int t   = i >> 5;
  const int hh  = t % (NHEADS + NKV);
  const int tok = t / (NHEADS + NKV);
  const int s   = tok & (SEQ - 1);
  const size_t base = (size_t)tok * QKVN +
                      (hh < NHEADS ? hh * HD : KOFF + (hh - NHEADS) * HD);
  const float ang = (float)s * __powf(10000.0f, -(float)d * (1.0f / 32.0f));
  float sn, cs;
  __sincosf(ang, &sn, &cs);
  const float sc = (hh < NHEADS) ? 0.125f : 1.0f;
  const float x1 = (float)qkv[base + d];
  const float x2 = (float)qkv[base + d + 32];
  qkv[base + d]      = (bf16)((x1 * cs - x2 * sn) * sc);
  qkv[base + d + 32] = (bf16)((x2 * cs + x1 * sn) * sc);
}

// ---------------------------------------------------------------- GEMM: C[M,N] = A[M,K] * W[N,K]^T
// 128x128 block, 8 waves (4x2), each wave 32(M) x 64(N) = 2x4 WMMA tiles.
// Double-buffered LDS tiles filled by async global->LDS copies.

template <typename OutT>
__global__ __launch_bounds__(256) void k_gemm_xwT(
    const bf16* __restrict__ A, const bf16* __restrict__ W,
    OutT* __restrict__ C, int K, int N)
{
  __shared__ alignas(16) bf16 As[2][128][32];
  __shared__ alignas(16) bf16 Ws[2][128][32];

  const int tid  = threadIdx.x;
  const int lane = tid & 31;
  const int wave = tid >> 5;
  const int wm   = wave >> 1;   // 0..3
  const int wn   = wave & 1;    // 0..1
  const size_t bm = (size_t)blockIdx.y * 128;
  const size_t bn = (size_t)blockIdx.x * 128;

  v8f acc[2][4];
  #pragma unroll
  for (int i = 0; i < 2; ++i)
    #pragma unroll
    for (int j = 0; j < 4; ++j) acc[i][j] = zero8();

  auto issue = [&](int buf, int k0) {
    #pragma unroll
    for (int i = 0; i < 2; ++i) {
      const int idx = i * 256 + tid;       // 0..511
      const int r   = idx >> 2;            // 0..127
      const int c   = (idx & 3) << 3;      // 0,8,16,24
      cp_async_b128(&As[buf][r][c], A + (bm + r) * K + k0 + c);
      cp_async_b128(&Ws[buf][r][c], W + (bn + r) * K + k0 + c);
    }
  };

  issue(0, 0);

  #pragma unroll 1
  for (int k0 = 0; k0 < K; k0 += 32) {
    const int buf = (k0 >> 5) & 1;
    cp_async_wait();        // this wave's tile `buf` landed in LDS
    __syncthreads();        // all waves landed; all waves done reading buf^1
    if (k0 + 32 < K) issue(buf ^ 1, k0 + 32);   // overlap with WMMA below

    v16bf af[2], wf[4];
    #pragma unroll
    for (int i = 0; i < 2; ++i) af[i] = frag_a(&As[buf][wm * 32 + i * 16][0], 32);
    #pragma unroll
    for (int j = 0; j < 4; ++j) wf[j] = frag_b_nk(&Ws[buf][wn * 64 + j * 16][0], 32);
    #pragma unroll
    for (int i = 0; i < 2; ++i)
      #pragma unroll
      for (int j = 0; j < 4; ++j)
        acc[i][j] = wmma_bf16(af[i], wf[j], acc[i][j]);
  }

  const int hl  = lane >> 4;
  const int col = lane & 15;
  #pragma unroll
  for (int i = 0; i < 2; ++i)
    #pragma unroll
    for (int j = 0; j < 4; ++j)
      #pragma unroll
      for (int r = 0; r < 8; ++r) {
        const size_t row = bm + wm * 32 + i * 16 + r + 8 * hl;
        const size_t cix = bn + wn * 64 + j * 16 + col;
        C[row * (size_t)N + cix] = (OutT)acc[i][j][r];
      }
}

// ---------------------------------------------------------------- attention
// grid: (SEQ/128, NHEADS, BATCH); 256 thr = 8 waves; wave w handles 16 queries.
// Flash loop over key tiles of 32; double-buffered K/V staged via async copies.

__global__ __launch_bounds__(256) void k_attn(
    const bf16* __restrict__ qkv, bf16* __restrict__ ctx)
{
  __shared__ alignas(16) bf16 Kt[2][32][64];
  __shared__ alignas(16) bf16 Vt[2][32][64];
  __shared__ alignas(16) bf16 Pw[8][16][32];

  const int tid  = threadIdx.x;
  const int lane = tid & 31;
  const int wave = tid >> 5;
  const int h    = blockIdx.y;
  const int kvh  = h >> 2;                    // GQA group = 4
  const size_t tokbase = (size_t)blockIdx.z * SEQ;
  const int q0   = blockIdx.x * 128 + wave * 16;

  const int hl  = lane >> 4;
  const int col = lane & 15;

  // preload Q fragments (2 k-steps of 32 over HD=64); 1/sqrt(HD) folded in.
  v16bf qf[2];
  #pragma unroll
  for (int s = 0; s < 2; ++s) {
    const bf16* qsrc = qkv + (tokbase + q0 + (lane & 15)) * QKVN + h * HD + s * 32;
    #pragma unroll
    for (int j = 0; j < 8; ++j) {
      const int k = (hl << 4) + (j << 1);
      qf[s][2*j]   = qsrc[k];
      qf[s][2*j+1] = qsrc[k + 1];
    }
  }

  v8f o[4];
  #pragma unroll
  for (int nt = 0; nt < 4; ++nt) o[nt] = zero8();
  float m8[8], l8[8];
  #pragma unroll
  for (int r = 0; r < 8; ++r) { m8[r] = -1e30f; l8[r] = 0.0f; }

  auto issue_kv = [&](int buf, int kt) {
    const int r = tid >> 3;            // key row 0..31
    const int c = (tid & 7) << 3;      // 0..56
    const size_t rowb = (tokbase + kt + r) * QKVN;
    cp_async_b128(&Kt[buf][r][c], qkv + rowb + KOFF + kvh * HD + c);
    cp_async_b128(&Vt[buf][r][c], qkv + rowb + VOFF + kvh * HD + c);
  };

  issue_kv(0, 0);

  #pragma unroll 1
  for (int kt = 0; kt < SEQ; kt += 32) {
    const int buf = (kt >> 5) & 1;
    cp_async_wait();
    __syncthreads();
    if (kt + 32 < SEQ) issue_kv(buf ^ 1, kt + 32);

    // scores: 16 queries x 32 keys (two 16-key WMMA tiles)
    v8f s0 = zero8(), s1 = zero8();
    #pragma unroll
    for (int d = 0; d < 2; ++d) {
      v16bf kb0 = frag_b_nk(&Kt[buf][0][d * 32], 64);
      v16bf kb1 = frag_b_nk(&Kt[buf][16][d * 32], 64);
      s0 = wmma_bf16(qf[d], kb0, s0);
      s1 = wmma_bf16(qf[d], kb1, s1);
    }

    // online softmax; row = r + 8*hl, col = lane&15 (C-matrix layout).
    // Row reductions are 16-lane butterflies done entirely in VALU (DPP16).
    float fac[8];
    #pragma unroll
    for (int r = 0; r < 8; ++r) {
      const float mx = dpp16_fmax(fmaxf(s0[r], s1[r]));
      const float mn = fmaxf(m8[r], mx);
      const float f  = __expf(m8[r] - mn);
      const float p0 = __expf(s0[r] - mn);
      const float p1 = __expf(s1[r] - mn);
      const float sum = dpp16_fadd(p0 + p1);
      l8[r] = l8[r] * f + sum;
      m8[r] = mn;
      fac[r] = f;
      Pw[wave][r + 8 * hl][col]      = (bf16)p0;
      Pw[wave][r + 8 * hl][col + 16] = (bf16)p1;
    }

    #pragma unroll
    for (int nt = 0; nt < 4; ++nt)
      #pragma unroll
      for (int r = 0; r < 8; ++r) o[nt][r] *= fac[r];

    // O += P(16x32) * V(32x64); per-wave P region, same-wave DS ordering is in-order
    v16bf pa = frag_a(&Pw[wave][0][0], 32);
    #pragma unroll
    for (int nt = 0; nt < 4; ++nt) {
      v16bf vb = frag_b_kn(&Vt[buf][0][nt * 16], 64);
      o[nt] = wmma_bf16(pa, vb, o[nt]);
    }
  }

  // write context: ctx[token][h*64 + d]
  #pragma unroll
  for (int nt = 0; nt < 4; ++nt)
    #pragma unroll
    for (int r = 0; r < 8; ++r) {
      const float val = o[nt][r] / l8[r];
      const size_t tok = tokbase + q0 + r + 8 * hl;
      ctx[tok * HIDDEN + h * HD + nt * 16 + col] = (bf16)val;
    }
}

// ---------------------------------------------------------------- launch

extern "C" void kernel_launch(void* const* d_in, const int* in_sizes, int n_in,
                              void* d_out, int out_size, void* d_ws, size_t ws_size,
                              hipStream_t stream) {
  (void)in_sizes; (void)n_in; (void)out_size; (void)ws_size;
  const float* hs = (const float*)d_in[0];
  const float* wq = (const float*)d_in[1];
  const float* wk = (const float*)d_in[2];
  const float* wv = (const float*)d_in[3];
  const float* wo = (const float*)d_in[4];
  float* out = (float*)d_out;

  char* w = (char*)d_ws;
  bf16* Xbf  = (bf16*)(w + 0);          // 4096*2048  bf16 = 16,777,216 B
  bf16* Wqkv = (bf16*)(w + 16777216);   // 3072*2048  bf16 = 12,582,912 B
  bf16* Wobf = (bf16*)(w + 29360128);   // 2048*2048  bf16 =  8,388,608 B
  bf16* QKV  = (bf16*)(w + 37748736);   // 4096*3072  bf16 = 25,165,824 B
  bf16* Ctx  = (bf16*)(w + 62914560);   // 4096*2048  bf16 = 16,777,216 B (end 79,691,776)

  { const int n = NTOK * HIDDEN;
    k_cvt<<<(n + 255) / 256, 256, 0, stream>>>(hs, Xbf, n); }
  { const int n = HIDDEN * HIDDEN;
    k_cvt<<<(n + 255) / 256, 256, 0, stream>>>(wo, Wobf, n); }
  { const int n = QKVN * HIDDEN;
    k_cvt_qkv<<<(n + 255) / 256, 256, 0, stream>>>(wq, wk, wv, Wqkv); }

  // QKV = X (4096x2048) * Wqkv^T (3072x2048)
  k_gemm_xwT<bf16><<<dim3(QKVN / 128, NTOK / 128), 256, 0, stream>>>(
      Xbf, Wqkv, QKV, HIDDEN, QKVN);

  { const int n = NTOK * (NHEADS + NKV) * 32;
    k_rope<<<(n + 255) / 256, 256, 0, stream>>>(QKV); }

  k_attn<<<dim3(SEQ / 128, NHEADS, BATCH), 256, 0, stream>>>(QKV, Ctx);

  // out = Ctx (4096x2048) * Wo^T (2048x2048), fp32 output
  k_gemm_xwT<float><<<dim3(HIDDEN / 128, NTOK / 128), 256, 0, stream>>>(
      Ctx, Wobf, out, HIDDEN, HIDDEN);
}
